// Swin3DEncoder_52725018525698
// MI455X (gfx1250) — matrix operand order repaired and checked by reading, not compiled, not run on hardware
//
#include <hip/hip_runtime.h>
#include <math.h>

typedef __attribute__((ext_vector_type(16))) _Float16 v16h;
typedef __attribute__((ext_vector_type(8)))  _Float16 v8h;
typedef __attribute__((ext_vector_type(8)))  float    v8f;
typedef __attribute__((ext_vector_type(4)))  float    v4f;

#define CDIV(a,b) (((a)+(b)-1)/(b))

// ---------------------------------------------------------------------------
// Patch embed: (2,1,64,64,64) --conv 2^3 stride 2--> (2,48,32,32,32)
// ---------------------------------------------------------------------------
__global__ __launch_bounds__(256) void k_patch_embed(
    const float* __restrict__ in, const float* __restrict__ w,
    const float* __restrict__ b, float* __restrict__ out)
{
  const int S = 32, C = 48;
  int idx = blockIdx.x * 256 + threadIdx.x;
  int total = 2 * C * S * S * S;
  if (idx >= total) return;
  int x = idx % S, y = (idx / S) % S, z = (idx / (S * S)) % S;
  int o = (idx / (S * S * S)) % C, bb = idx / (C * S * S * S);
  const float* ip = in + (size_t)bb * 64 * 64 * 64;
  float s = 0.f;
#pragma unroll
  for (int dz = 0; dz < 2; dz++)
#pragma unroll
    for (int dy = 0; dy < 2; dy++)
#pragma unroll
      for (int dx = 0; dx < 2; dx++)
        s += ip[(size_t)((2 * z + dz) * 64 + (2 * y + dy)) * 64 + (2 * x + dx)] *
             w[o * 8 + dz * 4 + dy * 2 + dx];
  out[idx] = s + b[o];
}

// ---------------------------------------------------------------------------
// Direct 3x3x3 conv, pad 1, NCDHW, in_ch == out_ch == C
// ---------------------------------------------------------------------------
__global__ __launch_bounds__(256) void k_conv3(
    const float* __restrict__ in, const float* __restrict__ w,
    float* __restrict__ out, int C, int S)
{
  int S3 = S * S * S;
  long total = 2L * C * S3;
  long idx = (long)blockIdx.x * 256 + threadIdx.x;
  if (idx >= total) return;
  int x = (int)(idx % S), y = (int)((idx / S) % S), z = (int)((idx / (S * S)) % S);
  int o = (int)((idx / S3) % C), bb = (int)(idx / ((long)C * S3));
  float s = 0.f;
  for (int c = 0; c < C; c++) {
    const float* ip = in + (size_t)(bb * C + c) * S3;
    const float* wp = w + (size_t)(o * C + c) * 27;
#pragma unroll
    for (int dz = -1; dz <= 1; dz++) {
      int zz = z + dz; if ((unsigned)zz >= (unsigned)S) continue;
#pragma unroll
      for (int dy = -1; dy <= 1; dy++) {
        int yy = y + dy; if ((unsigned)yy >= (unsigned)S) continue;
#pragma unroll
        for (int dx = -1; dx <= 1; dx++) {
          int xx = x + dx; if ((unsigned)xx >= (unsigned)S) continue;
          s += ip[(size_t)(zz * S + yy) * S + xx] *
               wp[(dz + 1) * 9 + (dy + 1) * 3 + (dx + 1)];
        }
      }
    }
  }
  out[idx] = s;
}

// ---------------------------------------------------------------------------
// Instance norm over spatial (one block per (b,c)), optional residual, lrelu
// out = lrelu(inorm(in) [+ res])
// ---------------------------------------------------------------------------
__global__ __launch_bounds__(256) void k_inorm_lrelu(
    const float* __restrict__ in, const float* __restrict__ res,
    float* __restrict__ out, int S3)
{
  __shared__ float sh[256], sh2[256];
  __shared__ float s_mu, s_rs;
  size_t base = (size_t)blockIdx.x * S3;
  float sum = 0.f, sq = 0.f;
  for (int i = threadIdx.x; i < S3; i += 256) {
    float v = in[base + i]; sum += v; sq += v * v;
  }
  sh[threadIdx.x] = sum; sh2[threadIdx.x] = sq; __syncthreads();
  for (int o = 128; o > 0; o >>= 1) {
    if ((int)threadIdx.x < o) { sh[threadIdx.x] += sh[threadIdx.x + o]; sh2[threadIdx.x] += sh2[threadIdx.x + o]; }
    __syncthreads();
  }
  if (threadIdx.x == 0) {
    float mu = sh[0] / S3;
    float var = sh2[0] / S3 - mu * mu;
    s_mu = mu; s_rs = rsqrtf(var + 1e-5f);
  }
  __syncthreads();
  float mu = s_mu, rs = s_rs;
  for (int i = threadIdx.x; i < S3; i += 256) {
    float v = (in[base + i] - mu) * rs;
    if (res) v += res[base + i];
    out[base + i] = v >= 0.f ? v : 0.01f * v;
  }
}

// ---------------------------------------------------------------------------
// Layout transposes
// ---------------------------------------------------------------------------
__global__ __launch_bounds__(256) void k_nchw2ndhwc(
    const float* __restrict__ in, float* __restrict__ out, int C, int S3)
{
  long total = 2L * C * S3;
  long idx = (long)blockIdx.x * 256 + threadIdx.x;
  if (idx >= total) return;
  int p = (int)(idx % S3); int c = (int)((idx / S3) % C); int bb = (int)(idx / ((long)C * S3));
  out[((size_t)bb * S3 + p) * C + c] = in[idx];
}

__global__ __launch_bounds__(256) void k_ndhwc2nchw(
    const float* __restrict__ in, float* __restrict__ out, int C, int S3)
{
  long total = 2L * C * S3;
  long idx = (long)blockIdx.x * 256 + threadIdx.x;
  if (idx >= total) return;
  int c = (int)(idx % C); int p = (int)((idx / C) % S3); int bb = (int)(idx / ((long)C * S3));
  out[((size_t)bb * C + c) * S3 + p] = in[idx];
}

// ---------------------------------------------------------------------------
// LayerNorm per row of C elements (g/b nullable)
// ---------------------------------------------------------------------------
__global__ __launch_bounds__(128) void k_layernorm(
    const float* __restrict__ in, const float* __restrict__ g,
    const float* __restrict__ b, float* __restrict__ out, int C)
{
  __shared__ float sh[128], sh2[128];
  __shared__ float s_mu, s_rs;
  size_t base = (size_t)blockIdx.x * C;
  float sum = 0.f, sq = 0.f;
  for (int i = threadIdx.x; i < C; i += 128) {
    float v = in[base + i]; sum += v; sq += v * v;
  }
  sh[threadIdx.x] = sum; sh2[threadIdx.x] = sq; __syncthreads();
  for (int o = 64; o > 0; o >>= 1) {
    if ((int)threadIdx.x < o) { sh[threadIdx.x] += sh[threadIdx.x + o]; sh2[threadIdx.x] += sh2[threadIdx.x + o]; }
    __syncthreads();
  }
  if (threadIdx.x == 0) {
    float mu = sh[0] / C;
    float var = sh2[0] / C - mu * mu;
    s_mu = mu; s_rs = rsqrtf(var + 1e-5f);
  }
  __syncthreads();
  float mu = s_mu, rs = s_rs;
  for (int i = threadIdx.x; i < C; i += 128) {
    float v = (in[base + i] - mu) * rs;
    if (g) v = v * g[i] + b[i];
    out[base + i] = v;
  }
}

// ---------------------------------------------------------------------------
// Window partition with pad + negative roll (shift).  xin: (2,S,S,S,C) NDHWC.
// xw: (2*nW3, N, C); pad tokens -> 0.
// ---------------------------------------------------------------------------
__global__ __launch_bounds__(256) void k_win_part(
    const float* __restrict__ xin, float* __restrict__ xw,
    int S, int dp, int ws, int ss, int nWa, int N, int C)
{
  int nW3 = nWa * nWa * nWa;
  long total = 2L * nW3 * N * C;
  long idx = (long)blockIdx.x * 256 + threadIdx.x;
  if (idx >= total) return;
  int c = (int)(idx % C);
  long t = idx / C;
  int i = (int)(t % N);
  long wg = t / N;
  int bb = (int)(wg / nW3); int wl = (int)(wg % nW3);
  int wz = wl / (nWa * nWa), wy = (wl / nWa) % nWa, wx = wl % nWa;
  int iz = i / (ws * ws), iy = (i / ws) % ws, ix = i % ws;
  int z = wz * ws + iz, y = wy * ws + iy, x = wx * ws + ix;
  int sz = (z + ss) % dp, sy = (y + ss) % dp, sx = (x + ss) % dp;
  float v = 0.f;
  if (sz < S && sy < S && sx < S)
    v = xin[(((size_t)bb * S + sz) * S + sy) * S * (size_t)C + (size_t)sx * C + c];
  xw[idx] = v;
}

// ---------------------------------------------------------------------------
// Window reverse + un-roll + crop + residual:  xio += proj(mapped)
// ---------------------------------------------------------------------------
__global__ __launch_bounds__(256) void k_win_rev(
    const float* __restrict__ proj, float* __restrict__ xio,
    int S, int dp, int ws, int ss, int nWa, int N, int C)
{
  int S3 = S * S * S;
  int nW3 = nWa * nWa * nWa;
  long total = 2L * S3 * C;
  long idx = (long)blockIdx.x * 256 + threadIdx.x;
  if (idx >= total) return;
  int c = (int)(idx % C);
  long t = idx / C;
  int p = (int)(t % S3); int bb = (int)(t / S3);
  int z = p / (S * S), y = (p / S) % S, x = p % S;
  int zr = (z - ss + dp) % dp, yr = (y - ss + dp) % dp, xr = (x - ss + dp) % dp;
  int wz = zr / ws, wy = yr / ws, wx = xr / ws;
  int iz = zr % ws, iy = yr % ws, ix = xr % ws;
  int wl = (wz * nWa + wy) * nWa + wx;
  long wg = (long)bb * nW3 + wl;
  int i = (iz * ws + iy) * ws + ix;
  xio[idx] += proj[((size_t)wg * N + i) * C + c];
}

// ---------------------------------------------------------------------------
// Windowed attention, one block per (window, head).  hd == 16 always.
// Relative-position bias from the 7x7x7 grid formula (matches REL_IDX[:N,:N]).
// Shift mask recomputed analytically (region classification on rolled grid).
// K/V tiles live in LDS (<= 343*16 fp32 each).
// ---------------------------------------------------------------------------
__global__ __launch_bounds__(256) void k_attention(
    const float* __restrict__ qkv, float* __restrict__ out,
    const float* __restrict__ rpb,
    int H, int C, int N, int nW3, int nWa, int ws, int ss, int dp)
{
  __shared__ float ksh[343 * 16];
  __shared__ float vsh[343 * 16];
  int w = blockIdx.x / H;
  int h = blockIdx.x % H;
  int C3 = 3 * C;
  size_t tbase = (size_t)w * N;
  for (int e = threadIdx.x; e < N * 16; e += 256) {
    int i = e >> 4, d = e & 15;
    size_t q = (tbase + i) * C3;
    ksh[e] = qkv[q + C + h * 16 + d];
    vsh[e] = qkv[q + 2 * C + h * 16 + d];
  }
  __syncthreads();
  int wl = w % nW3;
  int wz = wl / (nWa * nWa), wy = (wl / nWa) % nWa, wx = wl % nWa;
  bool shifted = (ss > 0);
  for (int i = threadIdx.x; i < N; i += 256) {
    float qv[16];
    size_t qb = (tbase + i) * C3 + h * 16;
#pragma unroll
    for (int d = 0; d < 16; d++) qv[d] = qkv[qb + d] * 0.25f;  // hd^-0.5
    int zi = i / 49, yi = (i / 7) % 7, xi = i % 7;
    int regi = 0;
    if (shifted) {
      int az = wz * ws + zi, ay = wy * ws + yi, ax = wx * ws + xi;
      int cz = (az >= dp - 7) + (az >= dp - 3);
      int cy = (ay >= dp - 7) + (ay >= dp - 3);
      int cx = (ax >= dp - 7) + (ax >= dp - 3);
      regi = (cz * 3 + cy) * 3 + cx;
    }
    // pass 1: row max
    float m = -3.4e38f;
    for (int j = 0; j < N; j++) {
      const float* kp = &ksh[j * 16];
      float s = 0.f;
#pragma unroll
      for (int d = 0; d < 16; d++) s += qv[d] * kp[d];
      int zj = j / 49, yj = (j / 7) % 7, xj = j % 7;
      int rel = ((zi - zj + 6) * 13 + (yi - yj + 6)) * 13 + (xi - xj + 6);
      s += rpb[rel * H + h];
      if (shifted) {
        int az = wz * ws + zj, ay = wy * ws + yj, ax = wx * ws + xj;
        int cz = (az >= dp - 7) + (az >= dp - 3);
        int cy = (ay >= dp - 7) + (ay >= dp - 3);
        int cx = (ax >= dp - 7) + (ax >= dp - 3);
        if (((cz * 3 + cy) * 3 + cx) != regi) s -= 100.f;
      }
      m = fmaxf(m, s);
    }
    // pass 2: exp / accumulate V
    float acc[16];
#pragma unroll
    for (int d = 0; d < 16; d++) acc[d] = 0.f;
    float sum = 0.f;
    for (int j = 0; j < N; j++) {
      const float* kp = &ksh[j * 16];
      float s = 0.f;
#pragma unroll
      for (int d = 0; d < 16; d++) s += qv[d] * kp[d];
      int zj = j / 49, yj = (j / 7) % 7, xj = j % 7;
      int rel = ((zi - zj + 6) * 13 + (yi - yj + 6)) * 13 + (xi - xj + 6);
      s += rpb[rel * H + h];
      if (shifted) {
        int az = wz * ws + zj, ay = wy * ws + yj, ax = wx * ws + xj;
        int cz = (az >= dp - 7) + (az >= dp - 3);
        int cy = (ay >= dp - 7) + (ay >= dp - 3);
        int cx = (ax >= dp - 7) + (ax >= dp - 3);
        if (((cz * 3 + cy) * 3 + cx) != regi) s -= 100.f;
      }
      float e = __expf(s - m);
      sum += e;
      const float* vp = &vsh[j * 16];
#pragma unroll
      for (int d = 0; d < 16; d++) acc[d] += e * vp[d];
    }
    float inv = 1.f / sum;
    size_t ob = (tbase + i) * C + h * 16;
#pragma unroll
    for (int d = 0; d < 16; d++) out[ob + d] = acc[d] * inv;
  }
}

// ---------------------------------------------------------------------------
// WMMA GEMM:  out[M,N] = A[M,K] @ W[N,K]^T (+bias[N]) (+res[M,N]) (opt. GELU)
// fp32 in/out, f16 operands via LDS, v_wmma_f32_16x16x32_f16 core.
// 256 threads = 8 waves; tile 128x64; each wave: 16 rows x 4 col-blocks.
// Tile staging uses whole 128-bit segments: every K in this net is a multiple
// of 16 and all row starts are 16-B aligned, so segments are never partially
// in range -> clean global_load_b128 / ds_store_b128 bursts.
// ---------------------------------------------------------------------------
__global__ __launch_bounds__(256) void k_gemm_wmma(
    const float* __restrict__ A, const float* __restrict__ W,
    const float* __restrict__ bias, const float* __restrict__ res,
    float* __restrict__ out, int M, int N, int K, int gelu_flag)
{
  __shared__ _Float16 ash[128 * 32];  // [m][k], k contiguous
  __shared__ _Float16 bsh[64 * 32];   // [n][k], k contiguous
  int tid = threadIdx.x;
  int lane = tid & 31;
  int wv = tid >> 5;
  int lm = lane & 15;
  int hi = lane >> 4;
  int m0 = blockIdx.y * 128;
  int n0 = blockIdx.x * 64;

  // A staging: one 16-float segment per thread (128 rows x 2 segments)
  int ar  = tid >> 1;
  int akb = (tid & 1) * 16;
  // B staging: one 8-float segment per thread (64 rows x 4 segments)
  int br  = tid >> 2;
  int bkb = (tid & 3) * 8;

  v8f zero = {0.f, 0.f, 0.f, 0.f, 0.f, 0.f, 0.f, 0.f};
  v8f acc[4] = {zero, zero, zero, zero};

  for (int k0 = 0; k0 < K; k0 += 32) {
    // ---- A tile ----
    {
      int gm = m0 + ar, gk = k0 + akb;
      v8h h0 = {}, h1 = {};
      if (gm < M && gk < K) {  // whole 16-elem segment in range (K % 16 == 0)
        const v4f* ap4 = (const v4f*)(A + (size_t)gm * K + gk);
        v4f f0 = ap4[0], f1 = ap4[1], f2 = ap4[2], f3 = ap4[3];
#pragma unroll
        for (int q = 0; q < 4; q++) {
          h0[q]     = (_Float16)f0[q];
          h0[4 + q] = (_Float16)f1[q];
          h1[q]     = (_Float16)f2[q];
          h1[4 + q] = (_Float16)f3[q];
        }
      }
      *(v8h*)&ash[ar * 32 + akb]     = h0;
      *(v8h*)&ash[ar * 32 + akb + 8] = h1;
    }
    // ---- B tile ----
    {
      int gn = n0 + br, gk = k0 + bkb;
      v8h h0 = {};
      if (gn < N && gk < K) {  // whole 8-elem segment in range
        const v4f* bp4 = (const v4f*)(W + (size_t)gn * K + gk);
        v4f f0 = bp4[0], f1 = bp4[1];
#pragma unroll
        for (int q = 0; q < 4; q++) {
          h0[q]     = (_Float16)f0[q];
          h0[4 + q] = (_Float16)f1[q];
        }
      }
      *(v8h*)&bsh[br * 32 + bkb] = h0;
    }
    __syncthreads();

    // A fragment (ISA 7.12.2: lane L holds row m=L&15; K halves = base..base+7
    // and base+16..base+23, base = (L>=16)?8:0)
    const _Float16* ap = &ash[(wv * 16 + lm) * 32 + hi * 8];
    v8h alo = *(const v8h*)ap;
    v8h ahi = *(const v8h*)(ap + 16);
    v16h a = __builtin_shufflevector(alo, ahi, 0, 1, 2, 3, 4, 5, 6, 7, 8, 9, 10, 11, 12, 13, 14, 15);

#pragma unroll
    for (int cb = 0; cb < 4; ++cb) {
      // B fragment: lane L holds col n=L&15; K halves = kb..kb+7 and kb+16..kb+23, kb=(L>=16)?8:0
      const _Float16* bp = &bsh[(cb * 16 + lm) * 32 + hi * 8];
      v8h blo = *(const v8h*)bp;
      v8h bhi = *(const v8h*)(bp + 16);
      v16h b = __builtin_shufflevector(blo, bhi, 0, 1, 2, 3, 4, 5, 6, 7, 8, 9, 10, 11, 12, 13, 14, 15);
      acc[cb] = __builtin_amdgcn_wmma_f32_16x16x32_f16(
          false, a, false, b, (short)0, acc[cb], false, false);
    }
    __syncthreads();
  }

  // Epilogue: C/D layout -> m = r + (lane>=16)*8, n = lane&15
#pragma unroll
  for (int cb = 0; cb < 4; ++cb) {
    int gn = n0 + cb * 16 + lm;
    if (gn >= N) continue;
    float bv = bias ? bias[gn] : 0.f;
#pragma unroll
    for (int r = 0; r < 8; ++r) {
      int gm = m0 + wv * 16 + r + hi * 8;
      if (gm >= M) continue;
      float v = acc[cb][r] + bv;
      if (res) v += res[(size_t)gm * N + gn];
      if (gelu_flag) v = 0.5f * v * (1.f + erff(v * 0.70710678118654752f));
      out[(size_t)gm * N + gn] = v;
    }
  }
}

// ---------------------------------------------------------------------------
// Patch-merging gather: (2,S,S,S,C) -> (2*(S/2)^3, 8C), concat order i,j,k
// ---------------------------------------------------------------------------
__global__ __launch_bounds__(256) void k_merge_gather(
    const float* __restrict__ xin, float* __restrict__ out, int S, int C)
{
  int S2 = S / 2;
  int S23 = S2 * S2 * S2;
  int C8 = 8 * C;
  long total = 2L * S23 * C8;
  long idx = (long)blockIdx.x * 256 + threadIdx.x;
  if (idx >= total) return;
  int c8 = (int)(idx % C8);
  long t = idx / C8;
  int p = (int)(t % S23); int bb = (int)(t / S23);
  int z2 = p / (S2 * S2), y2 = (p / S2) % S2, x2 = p % S2;
  int blk = c8 / C, c = c8 % C;
  int i = (blk >> 2) & 1, j = (blk >> 1) & 1, k = blk & 1;
  out[idx] = xin[((((size_t)bb * S + (2 * z2 + i)) * S + (2 * y2 + j)) * S + (2 * x2 + k)) * C + c];
}

// ---------------------------------------------------------------------------
// Final: normalize each of 16 tokens over 768 channels; then mean over the 8
// spatial positions per batch.
// ---------------------------------------------------------------------------
__global__ __launch_bounds__(256) void k_final_norm(
    const float* __restrict__ in, float* __restrict__ tmp)
{
  __shared__ float sh[256], sh2[256];
  __shared__ float s_mu, s_rs;
  const int C = 768;
  int tk = blockIdx.x;       // 0..15
  int bb = tk >> 3, p = tk & 7;
  float sum = 0.f, sq = 0.f;
  for (int c = threadIdx.x; c < C; c += 256) {
    float v = in[((size_t)bb * C + c) * 8 + p];
    sum += v; sq += v * v;
  }
  sh[threadIdx.x] = sum; sh2[threadIdx.x] = sq; __syncthreads();
  for (int o = 128; o > 0; o >>= 1) {
    if ((int)threadIdx.x < o) { sh[threadIdx.x] += sh[threadIdx.x + o]; sh2[threadIdx.x] += sh2[threadIdx.x + o]; }
    __syncthreads();
  }
  if (threadIdx.x == 0) {
    float mu = sh[0] / C;
    float var = sh2[0] / C - mu * mu;
    s_mu = mu; s_rs = rsqrtf(var + 1e-5f);
  }
  __syncthreads();
  float mu = s_mu, rs = s_rs;
  for (int c = threadIdx.x; c < C; c += 256)
    tmp[(size_t)tk * C + c] = (in[((size_t)bb * C + c) * 8 + p] - mu) * rs;
}

__global__ __launch_bounds__(256) void k_final_mean(
    const float* __restrict__ tmp, float* __restrict__ out)
{
  int idx = blockIdx.x * 256 + threadIdx.x;
  if (idx >= 1536) return;
  int bb = idx / 768, c = idx % 768;
  float s = 0.f;
  for (int p = 0; p < 8; p++) s += tmp[((size_t)(bb * 8 + p)) * 768 + c];
  out[idx] = s * 0.125f;
}

// ---------------------------------------------------------------------------
// Host side
// ---------------------------------------------------------------------------
struct BlockP {
  const float *n1_g, *n1_b, *qkv_w, *qkv_b, *rpb, *proj_w, *proj_b;
  const float *n2_g, *n2_b, *fc1_w, *fc1_b, *fc2_w, *fc2_b;
};
struct StageP {
  const float *c1_w, *c2_w;
  BlockP blk[2];
  const float *m_g, *m_b, *red_w;
};

static void launch_gemm(const float* A, const float* Wm, const float* bias,
                        const float* res, float* out, long M, int N, int K,
                        int gelu, hipStream_t st)
{
  dim3 g(CDIV(N, 64), CDIV((int)M, 128));
  k_gemm_wmma<<<g, dim3(256), 0, st>>>(A, Wm, bias, res, out, (int)M, N, K, gelu);
}

extern "C" void kernel_launch(void* const* d_in, const int* in_sizes, int n_in,
                              void* d_out, int out_size, void* d_ws, size_t ws_size,
                              hipStream_t stream)
{
  (void)n_in; (void)out_size; (void)ws_size;
  auto F = [&](int i) { return (const float*)d_in[i]; };

  const float *x = nullptr, *pe_w = nullptr, *pe_b = nullptr;
  StageP st[4];

  // Detect pytree flatten order: insertion order (x first, 2*64^3 elems) vs
  // jax sorted-key order (params first; pe_b with 48 elems leads).
  bool insertion = (in_sizes[0] == 2 * 64 * 64 * 64);
  if (insertion) {
    x = F(0); pe_w = F(1); pe_b = F(2);
    int idx = 3;
    for (int s = 0; s < 4; s++) {
      st[s].c1_w = F(idx++); st[s].c2_w = F(idx++);
      for (int j = 0; j < 2; j++) {
        BlockP& bp = st[s].blk[j];
        bp.n1_g = F(idx++); bp.n1_b = F(idx++);
        bp.qkv_w = F(idx++); bp.qkv_b = F(idx++);
        bp.rpb = F(idx++);
        bp.proj_w = F(idx++); bp.proj_b = F(idx++);
        bp.n2_g = F(idx++); bp.n2_b = F(idx++);
        bp.fc1_w = F(idx++); bp.fc1_b = F(idx++);
        bp.fc2_w = F(idx++); bp.fc2_b = F(idx++);
      }
      st[s].m_g = F(idx++); st[s].m_b = F(idx++); st[s].red_w = F(idx++);
    }
  } else {
    pe_b = F(0); pe_w = F(1);
    int idx = 2;
    for (int s = 0; s < 4; s++) {
      for (int j = 0; j < 2; j++) {
        BlockP& bp = st[s].blk[j];
        bp.fc1_b = F(idx++); bp.fc1_w = F(idx++);
        bp.fc2_b = F(idx++); bp.fc2_w = F(idx++);
        bp.n1_b = F(idx++); bp.n1_g = F(idx++);
        bp.n2_b = F(idx++); bp.n2_g = F(idx++);
        bp.proj_b = F(idx++); bp.proj_w = F(idx++);
        bp.qkv_b = F(idx++); bp.qkv_w = F(idx++);
        bp.rpb = F(idx++);
      }
      st[s].c1_w = F(idx++); st[s].c2_w = F(idx++);
      st[s].m_b = F(idx++); st[s].m_g = F(idx++);
      st[s].red_w = F(idx++);
    }
    x = F(idx);
  }

  // Workspace layout (floats).  Total ~34.1M floats (~137 MB) — fits in the
  // MI455X's 192 MB L2, so the whole pipeline stays cache-resident.
  const size_t SZ_BIG = 3200000;   // NCDHW/NDHWC activation buffers (<= 3.15M)
  const size_t SZ_WIN = 4300000;   // windowed T*C buffers (<= 4.12M)
  float* ws = (float*)d_ws;
  float* bA = ws;                  // NCDHW activation
  float* bB = bA + SZ_BIG;         // conv scratch / final tmp
  float* bC = bB + SZ_BIG;         // NDHWC activation (residual stream)
  float* bD = bC + SZ_BIG;         // LN output / merge gather
  float* bE = bD + SZ_BIG;         // xw / attn out / merge out
  float* bG = bE + SZ_WIN;         // proj out
  float* bF = bG + SZ_WIN;         // qkv / fc1 (<= 12.6M)

  static const int HEADS[4] = {3, 6, 12, 24};

  // Patch embed
  {
    long total = 2L * 48 * 32 * 32 * 32;
    k_patch_embed<<<CDIV(total, 256), 256, 0, stream>>>(x, pe_w, pe_b, bA);
  }

  for (int i = 0; i < 4; i++) {
    int S = 32 >> i;
    int C = 48 << i;
    int H = HEADS[i];
    int S3 = S * S * S;
    long tokens = 2L * S3;

    // --- Res block (NCDHW) ---
    k_conv3<<<CDIV(2L * C * S3, 256), 256, 0, stream>>>(bA, st[i].c1_w, bB, C, S);
    k_inorm_lrelu<<<2 * C, 256, 0, stream>>>(bB, nullptr, bB, S3);
    k_conv3<<<CDIV(2L * C * S3, 256), 256, 0, stream>>>(bB, st[i].c2_w, bD, C, S);
    k_inorm_lrelu<<<2 * C, 256, 0, stream>>>(bD, bA, bA, S3);   // +residual

    // NCDHW -> NDHWC residual stream
    k_nchw2ndhwc<<<CDIV(2L * C * S3, 256), 256, 0, stream>>>(bA, bC, C, S3);

    int wsz = (S < 7) ? S : 7;
    int pad = (wsz - (S % wsz)) % wsz;
    int dp = S + pad;
    int nWa = dp / wsz;
    int nW3 = nWa * nWa * nWa;
    int N = wsz * wsz * wsz;
    long T = 2L * nW3 * N;
    int shiftAmt = (S > 7) ? 3 : 0;

    // --- Swin blocks ---
    for (int j = 0; j < 2; j++) {
      const BlockP& bp = st[i].blk[j];
      int ss = (j == 1) ? shiftAmt : 0;

      k_layernorm<<<(unsigned)tokens, 128, 0, stream>>>(bC, bp.n1_g, bp.n1_b, bD, C);
      k_win_part<<<CDIV(T * C, 256), 256, 0, stream>>>(bD, bE, S, dp, wsz, ss, nWa, N, C);
      launch_gemm(bE, bp.qkv_w, bp.qkv_b, nullptr, bF, T, 3 * C, C, 0, stream);
      k_attention<<<(unsigned)(2 * nW3 * H), 256, 0, stream>>>(
          bF, bE, bp.rpb, H, C, N, nW3, nWa, wsz, ss, dp);
      launch_gemm(bE, bp.proj_w, bp.proj_b, nullptr, bG, T, C, C, 0, stream);
      k_win_rev<<<CDIV(2L * S3 * C, 256), 256, 0, stream>>>(
          bG, bC, S, dp, wsz, ss, nWa, N, C);                    // bC += attn

      k_layernorm<<<(unsigned)tokens, 128, 0, stream>>>(bC, bp.n2_g, bp.n2_b, bD, C);
      launch_gemm(bD, bp.fc1_w, bp.fc1_b, nullptr, bF, tokens, 4 * C, C, 1, stream);
      launch_gemm(bF, bp.fc2_w, bp.fc2_b, bC, bC, tokens, C, 4 * C, 0, stream);  // +residual
    }

    // --- Patch merging ---
    int S2 = S / 2;
    long t8 = 2L * S2 * S2 * S2;
    int C8 = 8 * C;
    k_merge_gather<<<CDIV(t8 * C8, 256), 256, 0, stream>>>(bC, bD, S, C);
    k_layernorm<<<(unsigned)t8, 128, 0, stream>>>(bD, st[i].m_g, st[i].m_b, bD, C8);
    launch_gemm(bD, st[i].red_w, nullptr, nullptr, bE, t8, 2 * C, C8, 0, stream);
    k_ndhwc2nchw<<<CDIV(t8 * 2 * C, 256), 256, 0, stream>>>(bE, bA, 2 * C, S2 * S2 * S2);
  }

  // Final LN over channels + spatial mean -> (2, 768)
  k_final_norm<<<16, 256, 0, stream>>>(bA, bB);
  k_final_mean<<<CDIV(1536, 256), 256, 0, stream>>>(bB, (float*)d_out);
}
